// PureLSTM_29317446763040
// MI455X (gfx1250) — matrix operand order repaired and checked by reading, not compiled, not run on hardware
//
#include <hip/hip_runtime.h>

typedef __attribute__((ext_vector_type(16))) _Float16 v16h;
typedef __attribute__((ext_vector_type(8)))  float    v8f;
typedef __attribute__((ext_vector_type(4)))  float    v4f;

#define LSTM_H 16
#define LSTM_I 32
#define LSTM_B 4096
#define LSTM_T 512
#define WPB    4   // waves (16-column batch tiles) per block

#if defined(__has_builtin)
#if __has_builtin(__builtin_amdgcn_tanhf)
#define LSTM_HAS_HW_TANH 1
#endif
#endif

__device__ __forceinline__ float tanh_f(float v) {
#ifdef LSTM_HAS_HW_TANH
  return __builtin_amdgcn_tanhf(v);                 // v_tanh_f32
#else
  return __builtin_fmaf(
      2.0f, __builtin_amdgcn_rcpf(1.0f + __expf(-2.0f * v)), -1.0f);
#endif
}
__device__ __forceinline__ float sigmoid_f(float v) {
#ifdef LSTM_HAS_HW_TANH
  return __builtin_fmaf(0.5f, tanh_f(0.5f * v), 0.5f);
#else
  return __builtin_amdgcn_rcpf(1.0f + __expf(-v));
#endif
}

// Transposed formulation: per 16-batch tile, per timestep,
//   G^T[gate][batch] = W_ih[gate][:] * x^T  +  W_hh[gate][:] * h^T  + b
// A = weight tile (M=16 gates, K), loop-invariant registers.
// B = feature^T (K x N=16 batch): x gives K=0..31, h gives K=0..15 (padded).
// D (C/D layout): lane = batch column, VGPR = gate row -> the LSTM pointwise
// update is lane-local, and h^T -> next B fragment needs no cross-lane exchange.
__global__ __launch_bounds__(WPB * 32, 1) void lstm_fused_kernel(
    const float* __restrict__ x,      // [B, T, I]
    const float* __restrict__ W_ih,   // [4H, I]
    const float* __restrict__ W_hh,   // [4H, H]
    const float* __restrict__ b_ih,   // [4H]
    const float* __restrict__ b_hh,   // [4H]
    const float* __restrict__ fc_w,   // [1, H]
    const float* __restrict__ fc_b,   // [1]
    float* __restrict__ out)          // [B]
{
  const int lane = threadIdx.x & 31;
  const int wave = threadIdx.x >> 5;
  const int tile = blockIdx.x * WPB + wave;
  const int b0   = tile * 16;          // first batch row of this wave's tile
  const int n    = lane & 15;          // batch column / gate-row-in-tile
  const bool hi  = lane >= 16;
  const int koff = hi ? 8 : 0;         // A-fragment K base for this lane half

  // ---- Loop-invariant A fragments (weights) -------------------------------
  // 16-bit A layout (16xK): lane<16 row M=n holds K{0..7,16..23},
  //                         lane>=16 row M=n holds K{8..15,24..31}.
  v16h a_ih[4];   // W_ih gate tiles, K=32 full
  v16h a_hh[4];   // W_hh gate tiles, K=16 real, K16..31 = 0
  v8f  cbias[4];  // bias: depends on gate row M -> per-VGPR element
  #pragma unroll
  for (int g = 0; g < 4; ++g) {
    const int grow = g * 16 + n;                   // this lane's gate row
    const float* wr = W_ih + grow * LSTM_I + koff;
    v16h fa;
    #pragma unroll
    for (int j = 0; j < 8; ++j) fa[j]     = (_Float16)wr[j];       // K koff+j
    #pragma unroll
    for (int j = 0; j < 8; ++j) fa[8 + j] = (_Float16)wr[16 + j];  // K koff+16+j
    a_ih[g] = fa;

    const float* hr = W_hh + grow * LSTM_H + koff;
    v16h fh;
    #pragma unroll
    for (int j = 0; j < 8; ++j) fh[j]     = (_Float16)hr[j];       // K koff+j
    #pragma unroll
    for (int j = 0; j < 8; ++j) fh[8 + j] = (_Float16)0.0f;        // K>=16 pad
    a_hh[g] = fh;

    v8f cb;
    #pragma unroll
    for (int j = 0; j < 8; ++j) {
      const int r = g * 16 + (hi ? 8 : 0) + j;     // gate row of VGPR elem j
      cb[j] = b_ih[r] + b_hh[r];
    }
    cbias[g] = cb;
  }

  // ---- State (h^T / c^T in C/D layout) ------------------------------------
  // lane n (either half) = batch column n; element j = hidden (hi?8:0)+j
  v8f hfrag = {};
  v8f cfrag = {};
  v16h bh = {};   // h^T B-fragment for the next step (starts at h0 = 0)

  // This lane's x column source: batch row n of the tile.
  const float* xrow = x + (size_t)(b0 + n) * LSTM_T * LSTM_I + (hi ? 8 : 0);

  for (int t = 0; t < LSTM_T; ++t) {
    const float* xp = xrow + (size_t)t * LSTM_I;
    if (t + 2 < LSTM_T) __builtin_prefetch(xp + 2 * LSTM_I, 0, 0);

    // ---- B-fragment of x^T (32 x 16): lane<16 K{0..7,16..23}, lane>=16 K{8..15,24..31} --
    // x is streamed exactly once -> non-temporal loads (bypass cache rinse).
    const v4f x0 = __builtin_nontemporal_load((const v4f*)(xp));
    const v4f x1 = __builtin_nontemporal_load((const v4f*)(xp + 4));
    const v4f x2 = __builtin_nontemporal_load((const v4f*)(xp + 16));
    const v4f x3 = __builtin_nontemporal_load((const v4f*)(xp + 20));
    v16h bx;
    #pragma unroll
    for (int j = 0; j < 4; ++j) {
      bx[j]      = (_Float16)x0[j];
      bx[4 + j]  = (_Float16)x1[j];
      bx[8 + j]  = (_Float16)x2[j];
      bx[12 + j] = (_Float16)x3[j];
    }

    // ---- Gates: 4 tiles x (x-WMMA + h-WMMA), f32 accumulate ---------------
    v8f acc[4];
    #pragma unroll
    for (int g = 0; g < 4; ++g) {
      acc[g] = __builtin_amdgcn_wmma_f32_16x16x32_f16(
          false, a_ih[g], false, bx, (short)0, cbias[g], false, false);
      acc[g] = __builtin_amdgcn_wmma_f32_16x16x32_f16(
          false, a_hh[g], false, bh, (short)0, acc[g], false, false);
    }

    // ---- Pointwise LSTM cell update (lane-local, f32) ---------------------
    #pragma unroll
    for (int j = 0; j < 8; ++j) {
      const float ig = sigmoid_f(acc[0][j]);
      const float fg = sigmoid_f(acc[1][j]);
      const float gg = tanh_f(acc[2][j]);
      const float og = sigmoid_f(acc[3][j]);
      const float cv = __builtin_fmaf(fg, cfrag[j], ig * gg);
      cfrag[j] = cv;
      hfrag[j] = og * tanh_f(cv);
    }

    // ---- Build next h^T B-fragment: lane<16 needs hidden 0..7 of batch n.
    // Own half supplies all 8 values; lane>=16 likewise holds its own hidden 8..15.
    union { _Float16 h[8]; int i[4]; } own;
    #pragma unroll
    for (int j = 0; j < 8; ++j) own.h[j] = (_Float16)hfrag[j];
    union { int i[4]; _Float16 h[8]; } oth;
    #pragma unroll
    for (int q = 0; q < 4; ++q) oth.i[q] = __shfl_xor(own.i[q], 16, 32);
    // elems 0..7 = hidden (hi?8:0)+j (own); elems 8..15 = K>=16 pad -> zero
    #pragma unroll
    for (int j = 0; j < 8; ++j) {
      bh[j]     = own.h[j];
      bh[8 + j] = (_Float16)0.0f;
    }
  }

  // ---- Final FC: out[b] = sigmoid(dot(h[b,:], fc_w) + fc_b) ---------------
  // lane holds hidden (hi?8:0)+j for batch n; partner holds the other half.
  float v = 0.0f;
  #pragma unroll
  for (int j = 0; j < 8; ++j)
    v = __builtin_fmaf(hfrag[j], fc_w[(hi ? 8 : 0) + j], v);
  v += __shfl_xor(v, 16, 32);
  if (!hi) out[b0 + n] = sigmoid_f(v + fc_b[0]);
}

extern "C" void kernel_launch(void* const* d_in, const int* in_sizes, int n_in,
                              void* d_out, int out_size, void* d_ws, size_t ws_size,
                              hipStream_t stream) {
  (void)in_sizes; (void)n_in; (void)d_ws; (void)ws_size; (void)out_size;
  const float* x    = (const float*)d_in[0];
  const float* W_ih = (const float*)d_in[1];
  const float* W_hh = (const float*)d_in[2];
  const float* b_ih = (const float*)d_in[3];
  const float* b_hh = (const float*)d_in[4];
  const float* fc_w = (const float*)d_in[5];
  const float* fc_b = (const float*)d_in[6];
  float* out = (float*)d_out;

  const int tiles  = LSTM_B / 16;        // 256 independent waves
  const int blocks = tiles / WPB;        // 64 blocks of 128 threads
  lstm_fused_kernel<<<blocks, WPB * 32, 0, stream>>>(
      x, W_ih, W_hh, b_ih, b_hh, fc_w, fc_b, out);
}
